// MoeMlp_64699387347314
// MI455X (gfx1250) — matrix operand order repaired and checked by reading, not compiled, not run on hardware
//
#include <hip/hip_runtime.h>
#include <hip/hip_bf16.h>

typedef __attribute__((ext_vector_type(16))) __bf16 v16bf;
typedef __attribute__((ext_vector_type(8)))  float  v8f;

// ---- problem dims (fixed by reference) ----
constexpr int T = 2048, H = 2048, I = 1408, E = 8;
constexpr int TWOI = 2 * I;
constexpr int NA = T * 2;           // 4096 assignments (top_k = 2)
constexpr int MT = 32;              // token rows per GEMM block
constexpr int PAD_CAP = 4352;       // 4096 + 8*31 headroom, padded per expert to 32

// ---- workspace layout ----
constexpr size_t align256(size_t x) { return (x + 255) & ~size_t(255); }
constexpr size_t WS_TOPK_IDX = 0;                                        // int   [NA]
constexpr size_t WS_TOPK_W   = align256(WS_TOPK_IDX + NA * 4);           // float [NA]
constexpr size_t WS_POS      = align256(WS_TOPK_W   + NA * 4);           // int   [NA]
constexpr size_t WS_STOK     = align256(WS_POS      + NA * 4);           // int   [PAD_CAP]
constexpr size_t WS_SW       = align256(WS_STOK     + PAD_CAP * 4);      // float [PAD_CAP]
constexpr size_t WS_OFF      = align256(WS_SW       + PAD_CAP * 4);      // int   [E+1]
constexpr size_t WS_SWISH    = align256(WS_OFF      + (E + 1) * 4);      // bf16  [PAD_CAP*I]
constexpr size_t WS_DRES     = align256(WS_SWISH + (size_t)PAD_CAP * I * 2); // float [PAD_CAP*H]

// LDS tile strides in uint (packed bf16 pair) units
constexpr int AS2 = 18;   // A tile: MT rows x 16 k-pairs
constexpr int BS2 = 18;   // B tile: 64 cols x 16 k-pairs (stored [n][kpair])

// single-instruction packed convert (mnemonic verified in prior disasm)
__device__ inline unsigned cvt_pk_bf16(float a, float b) {
    unsigned r;
    asm("v_cvt_pk_bf16_f32 %0, %1, %2" : "=v"(r) : "v"(a), "v"(b));
    return r;
}
__device__ inline float fast_silu_mul(float g, float u) {
    float sig = __builtin_amdgcn_rcpf(1.f + __expf(-g));   // v_rcp_f32
    return g * sig * u;
}

// ---------------- 1) router: logits -> softmax -> top2 -> renorm ----------------
__global__ __launch_bounds__(128) void k_router(const float* __restrict__ hs,
                                                const float* __restrict__ rw,
                                                int* __restrict__ topk_idx,
                                                float* __restrict__ topk_w) {
    int wave = threadIdx.x >> 5, lane = threadIdx.x & 31;
    int t = blockIdx.x * 4 + wave;
    float acc[E];
#pragma unroll
    for (int e = 0; e < E; e++) acc[e] = 0.f;
    const float* row = hs + (size_t)t * H;
    for (int h = lane; h < H; h += 32) {
        float x = row[h];
        const float* w = rw + (size_t)h * E;
#pragma unroll
        for (int e = 0; e < E; e++) acc[e] += x * w[e];
    }
#pragma unroll
    for (int e = 0; e < E; e++)
        for (int off = 16; off > 0; off >>= 1)
            acc[e] += __shfl_xor(acc[e], off, 32);
    if (lane == 0) {
        float m = acc[0];
        for (int e = 1; e < E; e++) m = fmaxf(m, acc[e]);
        float p[E], s = 0.f;
        for (int e = 0; e < E; e++) { p[e] = __expf(acc[e] - m); s += p[e]; }
        float invs = __builtin_amdgcn_rcpf(s);
        for (int e = 0; e < E; e++) p[e] *= invs;
        int i0 = 0;
        for (int e = 1; e < E; e++) if (p[e] > p[i0]) i0 = e;
        int i1 = (i0 == 0) ? 1 : 0;
        for (int e = 0; e < E; e++) if (e != i0 && p[e] > p[i1]) i1 = e;
        float w0 = p[i0], w1 = p[i1], inv = __builtin_amdgcn_rcpf(w0 + w1);
        topk_idx[t * 2 + 0] = i0; topk_idx[t * 2 + 1] = i1;
        topk_w[t * 2 + 0] = w0 * inv; topk_w[t * 2 + 1] = w1 * inv;
    }
}

// ---------------- 2) deterministic counting sort of assignments by expert ----------------
__global__ void k_sort(const int* __restrict__ topk_idx, const float* __restrict__ topk_w,
                       int* __restrict__ pos, int* __restrict__ stok,
                       float* __restrict__ sw, int* __restrict__ eoff) {
    if (threadIdx.x != 0 || blockIdx.x != 0) return;
    int cnt[E];
    for (int e = 0; e < E; e++) cnt[e] = 0;
    for (int a = 0; a < NA; a++) cnt[topk_idx[a]]++;
    int base[E + 1]; base[0] = 0;
    for (int e = 0; e < E; e++) base[e + 1] = base[e] + ((cnt[e] + MT - 1) & ~(MT - 1));
    for (int e = 0; e <= E; e++) eoff[e] = base[e];
    for (int s = 0; s < PAD_CAP; s++) { stok[s] = -1; sw[s] = 0.f; }
    int cur[E];
    for (int e = 0; e < E; e++) cur[e] = base[e];
    for (int a = 0; a < NA; a++) {
        int e = topk_idx[a];
        int p = cur[e]++;
        pos[a] = p;
        stok[p] = a >> 1;       // token id
        sw[p]  = topk_w[a];     // routing weight (pads stay 0)
    }
}

// ---- fragment builders (per ISA 16-bit A 16x32 / B 32x16 wave32 layouts) ----
struct Frag { union { v16bf v; unsigned u[8]; }; };

__device__ inline void load_afrag(Frag& a, const unsigned* __restrict__ At32,
                                  int rowbase, int m, int half) {
    const unsigned* Ar = &At32[(rowbase + m) * AS2];
    int k0 = half * 4;
#pragma unroll
    for (int j = 0; j < 4; j++) { a.u[j] = Ar[k0 + j]; a.u[4 + j] = Ar[8 + k0 + j]; }
}
__device__ inline void load_bfrag(Frag& b, const unsigned* __restrict__ Bt32,
                                  int col, int half) {
    const unsigned* Br = &Bt32[col * BS2];
    int k0 = half * 8;
#pragma unroll
    for (int j = 0; j < 8; j++) b.u[j] = Br[(j < 4 ? 0 : 4) + (k0 >> 1) + j];
}

// ---------------- 3) gate_up GEMM + SwiGLU (bf16 WMMA, f32 acc, M=32 N=64) ----------------
__global__ __launch_bounds__(128) void k_gateup(const float* __restrict__ hs,
                                                const float* __restrict__ guw,
                                                const int* __restrict__ stok,
                                                const int* __restrict__ eoff,
                                                __bf16* __restrict__ swish) {
    __shared__ unsigned At32[MT * AS2];
    __shared__ unsigned Bg32[64 * BS2];
    __shared__ unsigned Bu32[64 * BS2];
    __shared__ int tks[MT];
    __shared__ int s_e, s_total;

    int tid = threadIdx.x;
    int s0 = blockIdx.x * MT;
    if (tid == 0) {
        s_total = eoff[E];
        int e = 0;
        while (e < E && s0 >= eoff[e + 1]) e++;
        s_e = e;
    }
    __syncthreads();
    if (s0 >= s_total) return;            // uniform per block
    int expert = s_e;
    if (tid < MT) {
        int tok = stok[s0 + tid];
        tks[tid] = (tok < 0) ? 0 : tok;   // pads compute on token 0, results discarded
    }
    __syncthreads();

    int wave = tid >> 5, lane = tid & 31;
    int half = lane >> 4, n = lane & 15, nw = wave * 16;
    int i0 = blockIdx.y * 64;             // intermediate column base
    const float* gub = guw + (size_t)expert * H * TWOI;

    // ---- iteration-invariant per-thread staging offsets ----
    int    aLds[4]; size_t aGlob[4];
#pragma unroll
    for (int p4 = 0; p4 < 4; p4++) {
        int idx = tid + p4 * 128, r = idx >> 4, cp = idx & 15;
        aLds[p4]  = r * AS2 + cp;
        aGlob[p4] = (size_t)tks[r] * H + 2 * cp;
    }
    int    bLds[8]; size_t bGlob[8];
#pragma unroll
    for (int q = 0; q < 8; q++) {
        int idx = tid + q * 128, kp = idx >> 6, c = idx & 63;
        bLds[q]  = c * BS2 + kp;
        bGlob[q] = (size_t)(2 * kp) * TWOI + i0 + c;
    }

    v8f ag0 = {}, ag1 = {}, au0 = {}, au1 = {};

    // software pipeline: register-stage panel kb, overlap loads of kb+32 with WMMA(kb)
    float2 aR[4];
    float  bgA[8], bgB[8], buA[8], buB[8];
#pragma unroll
    for (int p4 = 0; p4 < 4; p4++) aR[p4] = *(const float2*)&hs[aGlob[p4]];
#pragma unroll
    for (int q = 0; q < 8; q++) {
        size_t gi = bGlob[q];
        bgA[q] = gub[gi];       bgB[q] = gub[gi + TWOI];
        buA[q] = gub[gi + I];   buB[q] = gub[gi + I + TWOI];
    }

    for (int kb = 0; kb < H; kb += 32) {
        __syncthreads();   // previous LDS reads complete
#pragma unroll
        for (int p4 = 0; p4 < 4; p4++) At32[aLds[p4]] = cvt_pk_bf16(aR[p4].x, aR[p4].y);
#pragma unroll
        for (int q = 0; q < 8; q++) {
            Bg32[bLds[q]] = cvt_pk_bf16(bgA[q], bgB[q]);
            Bu32[bLds[q]] = cvt_pk_bf16(buA[q], buB[q]);
        }
        __syncthreads();   // stores visible

        if (kb + 32 < H) { // issue next panel's global loads before compute
            int kn = kb + 32;
#pragma unroll
            for (int p4 = 0; p4 < 4; p4++) aR[p4] = *(const float2*)&hs[aGlob[p4] + kn];
            size_t kro = (size_t)kn * TWOI;
#pragma unroll
            for (int q = 0; q < 8; q++) {
                size_t gi = bGlob[q] + kro;
                bgA[q] = gub[gi];       bgB[q] = gub[gi + TWOI];
                buA[q] = gub[gi + I];   buB[q] = gub[gi + I + TWOI];
            }
        }

        Frag a0, a1, bg, bu;
        load_afrag(a0, At32, 0,  lane & 15, half);
        load_afrag(a1, At32, 16, lane & 15, half);
        load_bfrag(bg, Bg32, nw + n, half);
        load_bfrag(bu, Bu32, nw + n, half);

        ag0 = __builtin_amdgcn_wmma_f32_16x16x32_bf16(false, a0.v, false, bg.v, (short)0, ag0, false, false);
        ag1 = __builtin_amdgcn_wmma_f32_16x16x32_bf16(false, a1.v, false, bg.v, (short)0, ag1, false, false);
        au0 = __builtin_amdgcn_wmma_f32_16x16x32_bf16(false, a0.v, false, bu.v, (short)0, au0, false, false);
        au1 = __builtin_amdgcn_wmma_f32_16x16x32_bf16(false, a1.v, false, bu.v, (short)0, au1, false, false);
    }
    // SwiGLU epilogue; D layout: VGPR r -> row (r + 8*half), col n
#pragma unroll
    for (int r = 0; r < 8; r++) {
        int m = r + 8 * half;
        swish[(size_t)(s0 + m)      * I + i0 + nw + n] = (__bf16)fast_silu_mul(ag0[r], au0[r]);
        swish[(size_t)(s0 + 16 + m) * I + i0 + nw + n] = (__bf16)fast_silu_mul(ag1[r], au1[r]);
    }
}

// ---------------- 4) down GEMM, pre-scaled by routing weight (M=32 N=64) ----------------
__global__ __launch_bounds__(128) void k_down(const __bf16* __restrict__ swish,
                                              const float* __restrict__ dw,
                                              const int* __restrict__ eoff,
                                              const float* __restrict__ sw,
                                              float* __restrict__ dres) {
    __shared__ unsigned At32[MT * AS2];
    __shared__ unsigned Bt32[64 * BS2];
    __shared__ float wts[MT];
    __shared__ int s_e, s_total;

    int tid = threadIdx.x;
    int s0 = blockIdx.x * MT;
    if (tid == 0) {
        s_total = eoff[E];
        int e = 0;
        while (e < E && s0 >= eoff[e + 1]) e++;
        s_e = e;
    }
    __syncthreads();
    if (s0 >= s_total) return;
    int expert = s_e;
    if (tid < MT) wts[tid] = sw[s0 + tid];
    __syncthreads();

    int wave = tid >> 5, lane = tid & 31;
    int half = lane >> 4, n = lane & 15, nw = wave * 16;
    int h0 = blockIdx.y * 64;
    const float* db = dw + (size_t)expert * I * H;

    int    aLds[4]; size_t aGlob[4];
#pragma unroll
    for (int p4 = 0; p4 < 4; p4++) {
        int idx = tid + p4 * 128, r = idx >> 4, cp = idx & 15;
        aLds[p4]  = r * AS2 + cp;
        aGlob[p4] = (size_t)(s0 + r) * I + 2 * cp;   // bf16 units
    }
    int    bLds[8]; size_t bGlob[8];
#pragma unroll
    for (int q = 0; q < 8; q++) {
        int idx = tid + q * 128, kp = idx >> 6, c = idx & 63;
        bLds[q]  = c * BS2 + kp;
        bGlob[q] = (size_t)(2 * kp) * H + h0 + c;
    }

    v8f ac0 = {}, ac1 = {};

    unsigned aR[4];
    float    bA[8], bB[8];
#pragma unroll
    for (int p4 = 0; p4 < 4; p4++) aR[p4] = *(const unsigned*)&swish[aGlob[p4]];
#pragma unroll
    for (int q = 0; q < 8; q++) { bA[q] = db[bGlob[q]]; bB[q] = db[bGlob[q] + H]; }

    for (int kb = 0; kb < I; kb += 32) {
        __syncthreads();
#pragma unroll
        for (int p4 = 0; p4 < 4; p4++) At32[aLds[p4]] = aR[p4];
#pragma unroll
        for (int q = 0; q < 8; q++) Bt32[bLds[q]] = cvt_pk_bf16(bA[q], bB[q]);
        __syncthreads();

        if (kb + 32 < I) {
            int kn = kb + 32;
#pragma unroll
            for (int p4 = 0; p4 < 4; p4++) aR[p4] = *(const unsigned*)&swish[aGlob[p4] + kn];
            size_t kro = (size_t)kn * H;
#pragma unroll
            for (int q = 0; q < 8; q++) {
                bA[q] = db[bGlob[q] + kro];
                bB[q] = db[bGlob[q] + kro + H];
            }
        }

        Frag a0, a1, b;
        load_afrag(a0, At32, 0,  lane & 15, half);
        load_afrag(a1, At32, 16, lane & 15, half);
        load_bfrag(b, Bt32, nw + n, half);

        ac0 = __builtin_amdgcn_wmma_f32_16x16x32_bf16(false, a0.v, false, b.v, (short)0, ac0, false, false);
        ac1 = __builtin_amdgcn_wmma_f32_16x16x32_bf16(false, a1.v, false, b.v, (short)0, ac1, false, false);
    }
#pragma unroll
    for (int r = 0; r < 8; r++) {
        int m = r + 8 * half;
        dres[(size_t)(s0 + m)      * H + h0 + nw + n] = ac0[r] * wts[m];
        dres[(size_t)(s0 + 16 + m) * H + h0 + nw + n] = ac1[r] * wts[16 + m];
    }
}

// ---------------- 5) combine (no atomics, deterministic) ----------------
__global__ __launch_bounds__(256) void k_combine(const float* __restrict__ dres,
                                                 const int* __restrict__ pos,
                                                 float* __restrict__ out) {
    int idx = blockIdx.x * 256 + threadIdx.x;       // grid covers exactly T*H
    int t = idx / H, h = idx % H;
    out[idx] = dres[(size_t)pos[t * 2 + 0] * H + h] +
               dres[(size_t)pos[t * 2 + 1] * H + h];
}

extern "C" void kernel_launch(void* const* d_in, const int* in_sizes, int n_in,
                              void* d_out, int out_size, void* d_ws, size_t ws_size,
                              hipStream_t stream) {
    const float* hs  = (const float*)d_in[0];
    const float* rw  = (const float*)d_in[1];
    const float* guw = (const float*)d_in[2];
    const float* dw  = (const float*)d_in[3];
    // d_in[4] = top_k scalar (==2, hardcoded)

    char* ws = (char*)d_ws;
    int*     topk_idx = (int*)  (ws + WS_TOPK_IDX);
    float*   topk_w   = (float*)(ws + WS_TOPK_W);
    int*     pos      = (int*)  (ws + WS_POS);
    int*     stok     = (int*)  (ws + WS_STOK);
    float*   sw       = (float*)(ws + WS_SW);
    int*     eoff     = (int*)  (ws + WS_OFF);
    __bf16*  swish    = (__bf16*)(ws + WS_SWISH);
    float*   dres     = (float*)(ws + WS_DRES);

    k_router<<<T / 4, 128, 0, stream>>>(hs, rw, topk_idx, topk_w);
    k_sort<<<1, 1, 0, stream>>>(topk_idx, topk_w, pos, stok, sw, eoff);
    dim3 g1(PAD_CAP / MT, I / 64);
    k_gateup<<<g1, 128, 0, stream>>>(hs, guw, stok, eoff, swish);
    dim3 g2(PAD_CAP / MT, H / 64);
    k_down<<<g2, 128, 0, stream>>>(swish, dw, eoff, sw, dres);
    k_combine<<<(T * H) / 256, 256, 0, stream>>>(dres, pos, (float*)d_out);
}
